// GraphMIL_10892037063141
// MI455X (gfx1250) — matrix operand order repaired and checked by reading, not compiled, not run on hardware
//
#include <hip/hip_runtime.h>

#define N_NODES 50000
#define N_EDGES 800000
#define F_IN_   768
#define H_DIM   256
#define N_LAYERS 2
#define AH_     4
#define AD_     128
#define NC_     7
#define NBLK    64

typedef __attribute__((ext_vector_type(16))) __bf16 v16bf;
typedef __attribute__((ext_vector_type(8)))  __bf16 v8bf;
typedef __attribute__((ext_vector_type(4)))  __bf16 v4bf;
typedef __attribute__((ext_vector_type(8)))  float  v8f;

union V16 { v16bf v; v8bf h[2]; };

// ---------------------------------------------------------------------------
// staging helpers for the software-pipelined GEMM
// ---------------------------------------------------------------------------
__device__ __forceinline__ void stage_load(
    const float* __restrict__ A, const float* __restrict__ W,
    int M, int K, int Nc, int m0b, int n0b, int kc, int t,
    float4 av[4], float bv[8])
{
    #pragma unroll
    for (int i = 0; i < 4; ++i) {
        int idx = i * 256 + t;        // 0..1023 float4 slots of 128x32 tile
        int row = idx >> 3;
        int c4  = (idx & 7) << 2;
        int gr  = m0b + row;
        if (gr >= M) gr = M - 1;      // clamp: no divergence, rows never stored
        av[i] = *(const float4*)(A + (size_t)gr * K + kc + c4);
    }
    int col = t & 63;                 // B tile 32(K) x 64(N), read K-strided
    int kg  = (t >> 6) << 3;          // 0,8,16,24
    const float* wp = W + (size_t)(kc + kg) * Nc + n0b + col;
    #pragma unroll
    for (int j = 0; j < 8; ++j) bv[j] = wp[(size_t)j * Nc];
}

__device__ __forceinline__ void stage_store(
    __bf16 (*As)[40], __bf16 (*Bt)[40], int t,
    const float4 av[4], const float bv[8])
{
    #pragma unroll
    for (int i = 0; i < 4; ++i) {
        int idx = i * 256 + t;
        int row = idx >> 3;
        int c4  = (idx & 7) << 2;
        v4bf p;
        p[0] = (__bf16)av[i].x; p[1] = (__bf16)av[i].y;
        p[2] = (__bf16)av[i].z; p[3] = (__bf16)av[i].w;
        *(v4bf*)&As[row][c4] = p;
    }
    int col = t & 63;
    int kg  = (t >> 6) << 3;
    v8bf p;
    #pragma unroll
    for (int j = 0; j < 8; ++j) p[j] = (__bf16)bv[j];
    *(v8bf*)&Bt[col][kg] = p;         // transposed: Bt[col][k]
}

// ---------------------------------------------------------------------------
// GEMM: C[M,Nc] = A[M,K] @ W[K,Nc] + bias   (f32 in/out, bf16 WMMA compute)
// Block = 256 threads (8 waves). Block tile = 128(M) x 64(N), K-step = 32.
// Double-buffered LDS software pipeline: next tile's global loads are issued
// before the current tile's fragment reads + 4 WMMAs; convert+store after.
// A row-major, B transposed in LDS -> every fragment is 2 contiguous b128.
// ---------------------------------------------------------------------------
__global__ __launch_bounds__(256) void gemm_bias_wmma(
    const float* __restrict__ A, const float* __restrict__ W,
    const float* __restrict__ bias, float* __restrict__ C,
    int M, int K, int Nc)
{
    __shared__ __attribute__((aligned(16))) __bf16 As[2][128][40]; // 128x32
    __shared__ __attribute__((aligned(16))) __bf16 Bt[2][64][40];  // [col][k]

    const int t      = threadIdx.x;
    const int wave   = t >> 5;
    const int lane   = t & 31;
    const int lane16 = lane & 15;
    const int half   = lane >> 4;
    const int m0b    = blockIdx.x * 128;
    const int n0b    = blockIdx.y * 64;

    v8f acc[4];
    #pragma unroll
    for (int s = 0; s < 4; ++s) acc[s] = (v8f){0.f,0.f,0.f,0.f,0.f,0.f,0.f,0.f};

    // prologue: tile 0
    float4 av[4]; float bv[8];
    stage_load(A, W, M, K, Nc, m0b, n0b, 0, t, av, bv);
    stage_store(As[0], Bt[0], t, av, bv);

    const int nck = K >> 5;
    for (int ck = 0; ck < nck; ++ck) {
        __syncthreads();                          // stores of tile ck visible
        const int cur = ck & 1;
        const bool more = (ck + 1) < nck;         // uniform
        float4 av2[4]; float bv2[8];
        if (more)                                 // issue next tile's loads now
            stage_load(A, W, M, K, Nc, m0b, n0b, (ck + 1) << 5, t, av2, bv2);

        // ---- all fragment loads first (partial dscnt waits), then 4 WMMAs
        const int r = wave * 16 + lane16;
        V16 af;
        af.h[0] = *(const v8bf*)&As[cur][r][half * 8];       // k = 8h..8h+7
        af.h[1] = *(const v8bf*)&As[cur][r][16 + half * 8];  // k = 16+8h..+7
        V16 bf[4];
        #pragma unroll
        for (int s = 0; s < 4; ++s) {
            const int c = s * 16 + lane16;
            bf[s].h[0] = *(const v8bf*)&Bt[cur][c][half * 8];
            bf[s].h[1] = *(const v8bf*)&Bt[cur][c][16 + half * 8];
        }
        #pragma unroll
        for (int s = 0; s < 4; ++s)
            acc[s] = __builtin_amdgcn_wmma_f32_16x16x32_bf16(
                         false, af.v, false, bf[s].v, (short)0, acc[s], false, false);

        if (more)                                 // convert + store behind WMMAs
            stage_store(As[cur ^ 1], Bt[cur ^ 1], t, av2, bv2);
    }

    // ---- epilogue: C layout VGPR r -> M = r (lanes 0-15) / r+8 (lanes 16-31)
    #pragma unroll
    for (int s = 0; s < 4; ++s) {
        const int col = n0b + s * 16 + lane16;
        const float bia = bias[col];
        #pragma unroll
        for (int rI = 0; rI < 8; ++rI) {
            int row = m0b + wave * 16 + (half ? rI + 8 : rI);
            if (row < M) C[(size_t)row * Nc + col] = acc[s][rI] + bia;
        }
    }
}

// ---------------------------------------------------------------------------
// m = (1 + eps[layer]) * h
// ---------------------------------------------------------------------------
__global__ __launch_bounds__(256) void combine_init_kernel(
    const float* __restrict__ h, float* __restrict__ m,
    const float* __restrict__ eps, int layer, int n4)
{
    int i = blockIdx.x * 256 + threadIdx.x;
    if (i >= n4) return;
    float sc = 1.f + eps[layer];
    float4 v = ((const float4*)h)[i];
    v.x *= sc; v.y *= sc; v.z *= sc; v.w *= sc;
    ((float4*)m)[i] = v;
}

// ---------------------------------------------------------------------------
// m[dst] += h[src] over edges; thread = (edge, 4-channel chunk)
// ---------------------------------------------------------------------------
__global__ __launch_bounds__(256) void edge_scatter_kernel(
    const int* __restrict__ src, const int* __restrict__ dst,
    const float* __restrict__ h, float* __restrict__ m, int total)
{
    int gid = blockIdx.x * 256 + threadIdx.x;
    if (gid >= total) return;
    int e  = gid >> 6;            // H/4 = 64 chunks per edge
    int c4 = (gid & 63) << 2;
    int s = src[e], d = dst[e];
    float4 v = *(const float4*)(h + (size_t)s * H_DIM + c4);
    float* p = m + (size_t)d * H_DIM + c4;
    atomicAdd(p + 0, v.x);
    atomicAdd(p + 1, v.y);
    atomicAdd(p + 2, v.z);
    atomicAdd(p + 3, v.w);
}

// ---------------------------------------------------------------------------
// Y = relu(LN(X)*g+b) [+ res]  row-wise over D=256; one wave32 per row.
// ---------------------------------------------------------------------------
__global__ __launch_bounds__(256) void ln_relu_kernel(
    const float* __restrict__ X, const float* __restrict__ g, const float* __restrict__ b,
    const float* __restrict__ res, float* __restrict__ Y, int M)
{
    int wave = threadIdx.x >> 5, lane = threadIdx.x & 31;
    int row = blockIdx.x * 8 + wave;
    if (row >= M) return;
    const float* xr = X + (size_t)row * H_DIM;
    float v[8];
    float s = 0.f;
    #pragma unroll
    for (int j = 0; j < 8; ++j) { v[j] = xr[lane + 32*j]; s += v[j]; }
    #pragma unroll
    for (int o = 16; o; o >>= 1) s += __shfl_xor(s, o, 32);
    float mu = s * (1.f / 256.f);
    float q = 0.f;
    #pragma unroll
    for (int j = 0; j < 8; ++j) { float d = v[j] - mu; q += d * d; }
    #pragma unroll
    for (int o = 16; o; o >>= 1) q += __shfl_xor(q, o, 32);
    float inv = rsqrtf(q * (1.f / 256.f) + 1e-5f);
    float* yr = Y + (size_t)row * H_DIM;
    const float* rr = res ? res + (size_t)row * H_DIM : nullptr;
    #pragma unroll
    for (int j = 0; j < 8; ++j) {
        int c = lane + 32*j;
        float y = (v[j] - mu) * inv * g[c] + b[c];
        y = fmaxf(y, 0.f);
        if (rr) y += rr[c];
        yr[c] = y;
    }
}

// ---------------------------------------------------------------------------
// la[head,n] = tanh(h[n,:] @ Wa1[head] + ba1[head]) . wa2[head] + ba2[head]
// one wave per (node, head): 8 waves/block = 2 nodes x 4 heads
// ---------------------------------------------------------------------------
__global__ __launch_bounds__(256) void attn_score_kernel(
    const float* __restrict__ h, const float* __restrict__ Wa1, const float* __restrict__ ba1,
    const float* __restrict__ wa2, const float* __restrict__ ba2,
    float* __restrict__ la, int M)
{
    int wave = threadIdx.x >> 5, lane = threadIdx.x & 31;
    int node = blockIdx.x * 2 + (wave >> 2);
    int head = wave & 3;
    if (node >= M) return;
    float acc[4];
    #pragma unroll
    for (int j = 0; j < 4; ++j) acc[j] = ba1[head * AD_ + lane + 32*j];
    const float* hr = h + (size_t)node * H_DIM;
    const float* wb = Wa1 + (size_t)head * H_DIM * AD_;
    for (int d = 0; d < H_DIM; ++d) {
        float hv = hr[d];                    // wave-uniform broadcast load
        const float* wr = wb + d * AD_;
        #pragma unroll
        for (int j = 0; j < 4; ++j) acc[j] = fmaf(hv, wr[lane + 32*j], acc[j]);
    }
    float s = 0.f;
    #pragma unroll
    for (int j = 0; j < 4; ++j) s += tanhf(acc[j]) * wa2[head * AD_ + lane + 32*j];
    #pragma unroll
    for (int o = 16; o; o >>= 1) s += __shfl_xor(s, o, 32);
    if (lane == 0) la[(size_t)head * M + node] = s + ba2[head];
}

// ---------------------------------------------------------------------------
// softmax-over-N reductions (two stage)
// ---------------------------------------------------------------------------
__global__ __launch_bounds__(256) void la_max_partial_kernel(
    const float* __restrict__ la, float* __restrict__ pmax, int M)
{
    int head = blockIdx.y, blk = blockIdx.x;
    int chunk = (M + NBLK - 1) / NBLK;
    int start = blk * chunk;
    int end = start + chunk; if (end > M) end = M;
    float m = -3.4e38f;
    for (int i = start + threadIdx.x; i < end; i += 256)
        m = fmaxf(m, la[(size_t)head * M + i]);
    __shared__ float sm[256];
    sm[threadIdx.x] = m; __syncthreads();
    for (int s = 128; s; s >>= 1) {
        if (threadIdx.x < s) sm[threadIdx.x] = fmaxf(sm[threadIdx.x], sm[threadIdx.x + s]);
        __syncthreads();
    }
    if (threadIdx.x == 0) pmax[head * NBLK + blk] = sm[0];
}

__global__ __launch_bounds__(256) void la_expsum_partial_kernel(
    const float* __restrict__ la, const float* __restrict__ gmax,
    float* __restrict__ psum, int M)
{
    int head = blockIdx.y, blk = blockIdx.x;
    int chunk = (M + NBLK - 1) / NBLK;
    int start = blk * chunk;
    int end = start + chunk; if (end > M) end = M;
    float mx = gmax[head];
    float s = 0.f;
    for (int i = start + threadIdx.x; i < end; i += 256)
        s += __expf(la[(size_t)head * M + i] - mx);
    __shared__ float sm[256];
    sm[threadIdx.x] = s; __syncthreads();
    for (int st = 128; st; st >>= 1) {
        if (threadIdx.x < st) sm[threadIdx.x] += sm[threadIdx.x + st];
        __syncthreads();
    }
    if (threadIdx.x == 0) psum[head * NBLK + blk] = sm[0];
}

__global__ __launch_bounds__(64) void reduce_final_kernel(
    const float* __restrict__ part, float* __restrict__ out, int isMax)
{
    __shared__ float sm[64];
    int head = blockIdx.x, t = threadIdx.x;
    sm[t] = part[head * NBLK + t];
    __syncthreads();
    for (int s = 32; s; s >>= 1) {
        if (t < s) sm[t] = isMax ? fmaxf(sm[t], sm[t + s]) : (sm[t] + sm[t + s]);
        __syncthreads();
    }
    if (t == 0) out[head] = sm[0];
}

__global__ __launch_bounds__(256) void zero_f32_kernel(float* __restrict__ p, int n)
{
    int i = blockIdx.x * 256 + threadIdx.x;
    if (i < n) p[i] = 0.f;
}

// ---------------------------------------------------------------------------
// finalize: a = softmax(la); write attn[n,head]; z[head,:] += a * h[n,:]
// ---------------------------------------------------------------------------
__global__ __launch_bounds__(256) void attn_finalize_kernel(
    const float* __restrict__ la, const float* __restrict__ gmax,
    const float* __restrict__ gsum, const float* __restrict__ h,
    float* __restrict__ attn, float* __restrict__ z, int M)
{
    __shared__ float zsh[AH_ * H_DIM];
    int t = threadIdx.x;
    #pragma unroll
    for (int hd = 0; hd < AH_; ++hd) zsh[hd * H_DIM + t] = 0.f;
    __syncthreads();
    int base = blockIdx.x * 64;
    float mx[AH_], inv[AH_];
    #pragma unroll
    for (int hd = 0; hd < AH_; ++hd) { mx[hd] = gmax[hd]; inv[hd] = 1.f / gsum[hd]; }
    for (int i = 0; i < 64; ++i) {
        int n = base + i;
        if (n >= M) break;                       // uniform across block
        float hv = h[(size_t)n * H_DIM + t];
        #pragma unroll
        for (int hd = 0; hd < AH_; ++hd) {
            float a = __expf(la[(size_t)hd * M + n] - mx[hd]) * inv[hd];
            zsh[hd * H_DIM + t] += a * hv;
            if (t == hd) attn[(size_t)n * AH_ + hd] = a;
        }
    }
    __syncthreads();
    #pragma unroll
    for (int hd = 0; hd < AH_; ++hd)
        atomicAdd(&z[hd * H_DIM + t], zsh[hd * H_DIM + t]);
}

// ---------------------------------------------------------------------------
// classifier: z[4,256] -> probs[7]   (single block)
// ---------------------------------------------------------------------------
__global__ __launch_bounds__(256) void classifier_kernel(
    const float* __restrict__ z,
    const float* __restrict__ Wc1, const float* __restrict__ bc1,
    const float* __restrict__ gc1, const float* __restrict__ bec1,
    const float* __restrict__ Wc2, const float* __restrict__ bc2,
    const float* __restrict__ gc2, const float* __restrict__ bec2,
    const float* __restrict__ Wc3, const float* __restrict__ bc3,
    float* __restrict__ out)
{
    __shared__ float za[256];
    __shared__ float c1[128];
    __shared__ float c2[64];
    __shared__ float red[256];
    int t = threadIdx.x;
    za[t] = 0.25f * (z[t] + z[256 + t] + z[512 + t] + z[768 + t]);
    __syncthreads();

    float v1 = 0.f;
    if (t < 128) {
        float s = bc1[t];
        for (int d = 0; d < 256; ++d) s = fmaf(za[d], Wc1[d * 128 + t], s);
        v1 = s;
    }
    red[t] = (t < 128) ? v1 : 0.f;
    __syncthreads();
    for (int s = 128; s; s >>= 1) { if (t < s) red[t] += red[t + s]; __syncthreads(); }
    float mu = red[0] * (1.f / 128.f);
    __syncthreads();
    float dv = (t < 128) ? (v1 - mu) : 0.f;
    red[t] = dv * dv;
    __syncthreads();
    for (int s = 128; s; s >>= 1) { if (t < s) red[t] += red[t + s]; __syncthreads(); }
    float inv = rsqrtf(red[0] * (1.f / 128.f) + 1e-5f);
    __syncthreads();
    if (t < 128) c1[t] = fmaxf((v1 - mu) * inv * gc1[t] + bec1[t], 0.f);
    __syncthreads();

    float v2 = 0.f;
    if (t < 64) {
        float s = bc2[t];
        for (int d = 0; d < 128; ++d) s = fmaf(c1[d], Wc2[d * 64 + t], s);
        v2 = s;
    }
    red[t] = (t < 64) ? v2 : 0.f;
    __syncthreads();
    for (int s = 128; s; s >>= 1) { if (t < s) red[t] += red[t + s]; __syncthreads(); }
    float mu2 = red[0] * (1.f / 64.f);
    __syncthreads();
    float dv2 = (t < 64) ? (v2 - mu2) : 0.f;
    red[t] = dv2 * dv2;
    __syncthreads();
    for (int s = 128; s; s >>= 1) { if (t < s) red[t] += red[t + s]; __syncthreads(); }
    float inv2 = rsqrtf(red[0] * (1.f / 64.f) + 1e-5f);
    __syncthreads();
    if (t < 64) c2[t] = fmaxf((v2 - mu2) * inv2 * gc2[t] + bec2[t], 0.f);
    __syncthreads();

    if (t < NC_) {
        float s = bc3[t];
        for (int d = 0; d < 64; ++d) s = fmaf(c2[d], Wc3[d * NC_ + t], s);
        red[t] = s;
    }
    __syncthreads();
    if (t == 0) {
        float mxv = red[0];
        for (int c = 1; c < NC_; ++c) mxv = fmaxf(mxv, red[c]);
        float e[NC_]; float sum = 0.f;
        for (int c = 0; c < NC_; ++c) { e[c] = __expf(red[c] - mxv); sum += e[c]; }
        float invs = 1.f / sum;
        for (int c = 0; c < NC_; ++c) out[c] = e[c] * invs;
    }
}

// ---------------------------------------------------------------------------
extern "C" void kernel_launch(void* const* d_in, const int* in_sizes, int n_in,
                              void* d_out, int out_size, void* d_ws, size_t ws_size,
                              hipStream_t stream)
{
    (void)in_sizes; (void)n_in; (void)out_size; (void)ws_size;
    const float* x    = (const float*)d_in[0];
    const int*   ei   = (const int*)d_in[1];
    const float* Wp   = (const float*)d_in[2];
    const float* bp   = (const float*)d_in[3];
    const float* eps  = (const float*)d_in[4];
    const float* W1   = (const float*)d_in[5];
    const float* b1   = (const float*)d_in[6];
    const float* g1   = (const float*)d_in[7];
    const float* be1  = (const float*)d_in[8];
    const float* W2   = (const float*)d_in[9];
    const float* b2   = (const float*)d_in[10];
    const float* gl   = (const float*)d_in[11];
    const float* bl   = (const float*)d_in[12];
    const float* Wa1  = (const float*)d_in[13];
    const float* ba1  = (const float*)d_in[14];
    const float* wa2  = (const float*)d_in[15];
    const float* ba2  = (const float*)d_in[16];
    const float* Wc1  = (const float*)d_in[17];
    const float* bc1  = (const float*)d_in[18];
    const float* gc1  = (const float*)d_in[19];
    const float* bec1 = (const float*)d_in[20];
    const float* Wc2  = (const float*)d_in[21];
    const float* bc2  = (const float*)d_in[22];
    const float* gc2  = (const float*)d_in[23];
    const float* bec2 = (const float*)d_in[24];
    const float* Wc3  = (const float*)d_in[25];
    const float* bc3  = (const float*)d_in[26];

    const int* srcI = ei;
    const int* dstI = ei + N_EDGES;

    float* ws = (float*)d_ws;
    const size_t NH = (size_t)N_NODES * H_DIM;
    float* h0   = ws;
    float* h1   = h0 + NH;
    float* mB   = h1 + NH;
    float* uB   = mB + NH;
    float* la   = uB + NH;
    float* z    = la + (size_t)AH_ * N_NODES;
    float* gmax = z + AH_ * H_DIM;
    float* gsum = gmax + AH_;
    float* pmax = gsum + AH_;
    float* psum = pmax + AH_ * NBLK;

    float* out = (float*)d_out;

    dim3 gGemm((N_NODES + 127) / 128, (H_DIM + 63) / 64);

    // input projection: h0 = x @ Wp + bp
    gemm_bias_wmma<<<gGemm, 256, 0, stream>>>(x, Wp, bp, h0, N_NODES, F_IN_, H_DIM);

    for (int i = 0; i < N_LAYERS; ++i) {
        const float* hIn = (i == 0) ? h0 : h1;
        float* hOut      = (i == 0) ? h1 : h0;
        int n4 = (int)(NH / 4);
        combine_init_kernel<<<(n4 + 255) / 256, 256, 0, stream>>>(hIn, mB, eps, i, n4);
        int total = N_EDGES * (H_DIM / 4);
        edge_scatter_kernel<<<(total + 255) / 256, 256, 0, stream>>>(srcI, dstI, hIn, mB, total);
        gemm_bias_wmma<<<gGemm, 256, 0, stream>>>(mB, W1 + (size_t)i * H_DIM * H_DIM,
                                                  b1 + i * H_DIM, uB, N_NODES, H_DIM, H_DIM);
        ln_relu_kernel<<<(N_NODES + 7) / 8, 256, 0, stream>>>(uB, g1 + i * H_DIM,
                                                              be1 + i * H_DIM, nullptr, uB, N_NODES);
        gemm_bias_wmma<<<gGemm, 256, 0, stream>>>(uB, W2 + (size_t)i * H_DIM * H_DIM,
                                                  b2 + i * H_DIM, mB, N_NODES, H_DIM, H_DIM);
        ln_relu_kernel<<<(N_NODES + 7) / 8, 256, 0, stream>>>(mB, gl + i * H_DIM,
                                                              bl + i * H_DIM, hIn, hOut, N_NODES);
    }
    const float* hF = h0;   // layer 0 -> h1, layer 1 -> h0

    attn_score_kernel<<<(N_NODES + 1) / 2, 256, 0, stream>>>(hF, Wa1, ba1, wa2, ba2, la, N_NODES);
    la_max_partial_kernel<<<dim3(NBLK, AH_), 256, 0, stream>>>(la, pmax, N_NODES);
    reduce_final_kernel<<<AH_, 64, 0, stream>>>(pmax, gmax, 1);
    la_expsum_partial_kernel<<<dim3(NBLK, AH_), 256, 0, stream>>>(la, gmax, psum, N_NODES);
    reduce_final_kernel<<<AH_, 64, 0, stream>>>(psum, gsum, 0);
    zero_f32_kernel<<<(AH_ * H_DIM + 255) / 256, 256, 0, stream>>>(z, AH_ * H_DIM);
    attn_finalize_kernel<<<(N_NODES + 63) / 64, 256, 0, stream>>>(la, gmax, gsum, hF,
                                                                  out + NC_, z, N_NODES);
    classifier_kernel<<<1, 256, 0, stream>>>(z, Wc1, bc1, gc1, bec1,
                                             Wc2, bc2, gc2, bec2, Wc3, bc3, out);
}
